// KANConv2D_12558484373526
// MI455X (gfx1250) — matrix operand order repaired and checked by reading, not compiled, not run on hardware
//
#include <hip/hip_runtime.h>
#include <hip/hip_bf16.h>

// ---------------------------------------------------------------------------
// KAN Conv2D for MI455X (gfx1250, wave32, WMMA bf16 16x16x32)
//
// out = conv(x,kernel) + exp(-gamma * dist) + bias
// dist = ||patch||^2 - 2*(patch.cp) + ||cp||^2,  gamma = 1/(2*mean(dist))
//
// Implicit GEMM: M = 16*222*222 = 788544 pixels, K = 288 (9 taps x 32ch),
// N = 128 (kernel 0..63 || control_points 64..127).
//  - x and B pre-converted to bf16 once (prep kernels)
//  - per-tap 8KB B tiles streamed to LDS via GLOBAL_LOAD_ASYNC_TO_LDS_B128
//    (ASYNCcnt), double buffered, overlapping the 8 WMMAs per tap
//  - A fragments loaded straight from bf16 x (no in-loop conversion VALU)
// ---------------------------------------------------------------------------

typedef __attribute__((ext_vector_type(16))) __bf16 v16bf;
typedef __attribute__((ext_vector_type(8)))  float  v8f;

#define BATCH   16
#define INH     224
#define INW     224
#define CIN     32
#define OHW     222
#define NPIX    (BATCH * OHW * OHW)          // 788544
#define NFILT   64
#define KTAPS   9
#define NTOT    128
#define KSTRIDE 40                           // LDS row stride in bf16 (80 B)

#define MBLOCK      128
#define NBLK_GEMM   ((NPIX + MBLOCK - 1) / MBLOCK)     // 6161
#define NBLK_PSQ    ((NPIX + 255) / 256)               // 3081
#define TOTAL4      (NPIX * (NFILT / 4))
#define BTILE_ELEMS (NTOT * 32)              // 4096 bf16 per tap, [n][32]
#define BUF_ELEMS   (NTOT * KSTRIDE)         // LDS elems per buffer
#define NXEL        (BATCH * INH * INW * CIN)          // 25,690,112
#define NXUNITS     (NXEL / 8)                         // 3,211,264 (8 elems/thr)

__device__ __forceinline__ unsigned short f2bf_bits(float f) {
    union { float f; unsigned u; } v; v.f = f;
    unsigned u = v.u + 0x7FFFu + ((v.u >> 16) & 1u);   // round-to-nearest-even
    return (unsigned short)(u >> 16);
}

// Async global->LDS 16-byte copy (CDNA5 GLOBAL_LOAD_ASYNC_TO_LDS_B128).
// lds_off: byte offset within the workgroup LDS allocation (flat addr[31:0]).
__device__ __forceinline__ void async_copy_b128(unsigned lds_off, const void* g) {
    unsigned long long ga = (unsigned long long)(uintptr_t)g;
    asm volatile("global_load_async_to_lds_b128 %0, %1, off"
                 :: "v"(lds_off), "v"(ga)
                 : "memory");
}

__device__ __forceinline__ void wait_async0() {
#if __has_builtin(__builtin_amdgcn_s_wait_asynccnt)
    __builtin_amdgcn_s_wait_asynccnt(0);
#else
    asm volatile("s_wait_asynccnt 0x0" ::: "memory");
#endif
}

// ---------------------------------------------------------------------------
// Kernel 0a (one-time): x f32 -> bf16, same NHWC layout. 8 elems per thread.
// ---------------------------------------------------------------------------
__global__ __launch_bounds__(256) void kan_prepx_kernel(
    const float* __restrict__ x,
    unsigned short* __restrict__ xb)
{
    size_t i = (size_t)blockIdx.x * 256 + threadIdx.x;   // unit = 8 elements
    const float4* src = (const float4*)x;
    float4 f0 = src[i * 2];
    float4 f1 = src[i * 2 + 1];
    union { unsigned short t[8]; uint4 q; } u;
    u.t[0] = f2bf_bits(f0.x); u.t[1] = f2bf_bits(f0.y);
    u.t[2] = f2bf_bits(f0.z); u.t[3] = f2bf_bits(f0.w);
    u.t[4] = f2bf_bits(f1.x); u.t[5] = f2bf_bits(f1.y);
    u.t[6] = f2bf_bits(f1.z); u.t[7] = f2bf_bits(f1.w);
    ((uint4*)xb)[i] = u.q;
}

// ---------------------------------------------------------------------------
// Kernel 0b (one-time): B = [kernel || control_points] -> bf16,
// transposed to [tap][n][32ch] so the GEMM can DMA flat 16B chunks.
// ---------------------------------------------------------------------------
__global__ __launch_bounds__(256) void kan_prepb_kernel(
    const float* __restrict__ wconv,
    const float* __restrict__ wcp,
    unsigned short* __restrict__ bws)
{
    int e = blockIdx.x * 256 + threadIdx.x;          // 0 .. 9*4096-1
    if (e >= KTAPS * BTILE_ELEMS) return;
    int tap = e >> 12;
    int r   = e & 4095;
    int n   = r >> 5;          // wave-uniform (32 lanes share one n)
    int k   = r & 31;
    float v = (n < NFILT) ? wconv[(tap * 32 + k) * NFILT + n]
                          : wcp  [(tap * 32 + k) * NFILT + (n - NFILT)];
    bws[e] = f2bf_bits(v);
}

// ---------------------------------------------------------------------------
// Kernel 1: implicit-GEMM with WMMA bf16.  Block = 256 thr = 8 waves.
// Block tile 128M x 128N; wave (wm 0..3, wn 0..1) owns 2x4 accumulator tiles.
// ---------------------------------------------------------------------------
__global__ __launch_bounds__(256) void kan_gemm_kernel(
    const unsigned short* __restrict__ xb,    // bf16 x, NHWC
    const unsigned short* __restrict__ bws,   // bf16 B, [tap][n][32]
    float* __restrict__ out,
    float* __restrict__ cross,
    float* __restrict__ crossPart)
{
    __shared__ __align__(16) unsigned short Bs[2 * BUF_ELEMS];  // 2 x 10KB
    __shared__ float sred[256];

    const int tid  = threadIdx.x;
    const int lane = tid & 31;
    const int wave = tid >> 5;
    const int wm   = wave & 3;
    const int wn   = wave >> 2;               // 0 = conv half, 1 = cross half
    const int mBase = blockIdx.x * MBLOCK + wm * 32;
    const int nBase = wn * 64;

    const int lr = lane & 15;
    const int lh = lane >> 4;

    // LDS base byte offset of Bs (flat shared address truncates to LDS offset)
    const unsigned bsBase = (unsigned)(uintptr_t)(void*)&Bs[0];
    // Each thread DMAs two 16B chunks per tap: chunk q -> row q>>2, 16B slot q&3
    const int q0 = tid, q1 = tid + 256;       // of 512 chunks (8KB tile)
    const unsigned ldsOff0 = bsBase + (unsigned)((q0 >> 2) * (KSTRIDE * 2) + (q0 & 3) * 16);
    const unsigned ldsOff1 = bsBase + (unsigned)((q1 >> 2) * (KSTRIDE * 2) + (q1 & 3) * 16);

    // Per-lane source row pointers for the 2 M tiles (clamped for tail block)
    const unsigned short* rowPtr[2];
    #pragma unroll
    for (int mt = 0; mt < 2; ++mt) {
        int p  = mBase + mt * 16 + lr;
        int pc = (p < NPIX) ? p : (NPIX - 1);
        int b   = pc / (OHW * OHW);
        int rem = pc - b * (OHW * OHW);
        int oh  = rem / OHW;
        int ow  = rem - oh * OHW;
        rowPtr[mt] = xb + (((size_t)b * INH + oh) * INW + ow) * CIN;
    }

    v8f acc[2][4];
    #pragma unroll
    for (int mt = 0; mt < 2; ++mt)
        #pragma unroll
        for (int nt = 0; nt < 4; ++nt)
            #pragma unroll
            for (int v = 0; v < 8; ++v) acc[mt][nt][v] = 0.0f;

    const int chA = lh * 16;   // lane's A channels: chA..chA+15

    // ---- prologue: DMA tap 0 into buffer 0 ----
    {
        const unsigned short* src = bws;      // tap 0
        async_copy_b128(ldsOff0, src + q0 * 8);
        async_copy_b128(ldsOff1, src + q1 * 8);
    }

    for (int tap = 0; tap < KTAPS; ++tap) {
        wait_async0();
        __syncthreads();                      // buffer tap&1 ready for all waves

        // ---- DMA next tap into the other buffer (overlaps WMMA below) ----
        if (tap + 1 < KTAPS) {
            const unsigned short* src = bws + (tap + 1) * BTILE_ELEMS;
            unsigned bufB = (unsigned)(((tap + 1) & 1) * (BUF_ELEMS * 2));
            async_copy_b128(ldsOff0 + bufB, src + q0 * 8);
            async_copy_b128(ldsOff1 + bufB, src + q1 * 8);
        }

        const int kh = tap / 3, kw = tap - kh * 3;
        const int tapOff = (kh * INW + kw) * CIN;
        const unsigned short* bufBs = Bs + (tap & 1) * BUF_ELEMS;

        // ---- A fragments: two 16B bf16 loads per M tile, no conversion ----
        v16bf afrag[2];
        #pragma unroll
        for (int mt = 0; mt < 2; ++mt) {
            const unsigned short* p = rowPtr[mt] + tapOff + chA;
            __builtin_memcpy(&afrag[mt], __builtin_assume_aligned(p, 16), 16);
            __builtin_memcpy((char*)&afrag[mt] + 16,
                             __builtin_assume_aligned(p + 8, 16), 16);
        }

        // ---- 8 WMMAs ----
        #pragma unroll
        for (int nt = 0; nt < 4; ++nt) {
            const unsigned short* bp =
                bufBs + (nBase + nt * 16 + lr) * KSTRIDE + lh * 16;
            v16bf bfrag;
            __builtin_memcpy(&bfrag, __builtin_assume_aligned(bp, 16), 32);
            #pragma unroll
            for (int mt = 0; mt < 2; ++mt) {
                acc[mt][nt] = __builtin_amdgcn_wmma_f32_16x16x32_bf16(
                    false, afrag[mt], false, bfrag,
                    (short)0, acc[mt][nt], false, false);
            }
        }
        __syncthreads();   // all waves done with buffer tap&1 before reuse
    }

    // ---- epilogue: uniform-branch stores + deterministic cross sum ----
    float localSum = 0.0f;
    const bool fullBlock = (blockIdx.x != NBLK_GEMM - 1);
    if (wn == 0) {
        if (fullBlock) {
            #pragma unroll
            for (int mt = 0; mt < 2; ++mt)
                #pragma unroll
                for (int nt = 0; nt < 4; ++nt)
                    #pragma unroll
                    for (int v = 0; v < 8; ++v) {
                        size_t p = (size_t)(mBase + mt * 16 + lh * 8 + v);
                        out[p * NFILT + nt * 16 + lr] = acc[mt][nt][v];
                    }
        } else {
            #pragma unroll
            for (int mt = 0; mt < 2; ++mt)
                #pragma unroll
                for (int nt = 0; nt < 4; ++nt)
                    #pragma unroll
                    for (int v = 0; v < 8; ++v) {
                        int p = mBase + mt * 16 + lh * 8 + v;
                        if (p < NPIX)
                            out[(size_t)p * NFILT + nt * 16 + lr] = acc[mt][nt][v];
                    }
        }
    } else {
        if (fullBlock) {
            #pragma unroll
            for (int mt = 0; mt < 2; ++mt)
                #pragma unroll
                for (int nt = 0; nt < 4; ++nt)
                    #pragma unroll
                    for (int v = 0; v < 8; ++v) {
                        size_t p = (size_t)(mBase + mt * 16 + lh * 8 + v);
                        float val = acc[mt][nt][v];
                        cross[p * NFILT + nt * 16 + lr] = val;
                        localSum += val;
                    }
        } else {
            #pragma unroll
            for (int mt = 0; mt < 2; ++mt)
                #pragma unroll
                for (int nt = 0; nt < 4; ++nt)
                    #pragma unroll
                    for (int v = 0; v < 8; ++v) {
                        int p = mBase + mt * 16 + lh * 8 + v;
                        if (p < NPIX) {
                            float val = acc[mt][nt][v];
                            cross[(size_t)p * NFILT + nt * 16 + lr] = val;
                            localSum += val;
                        }
                    }
        }
    }

    sred[tid] = localSum;
    __syncthreads();
    #pragma unroll
    for (int s = 128; s > 0; s >>= 1) {
        if (tid < s) sred[tid] += sred[tid + s];
        __syncthreads();
    }
    if (tid == 0) crossPart[blockIdx.x] = sred[0];
}

// ---------------------------------------------------------------------------
// Kernel 2: ||patch||^2 per output pixel + per-block partial sums (f32 x).
// ---------------------------------------------------------------------------
__global__ __launch_bounds__(256) void kan_psq_kernel(
    const float* __restrict__ x,
    float* __restrict__ psq,
    float* __restrict__ psqPart)
{
    __shared__ float sr[256];
    int p = blockIdx.x * 256 + threadIdx.x;
    float s = 0.0f;
    if (p < NPIX) {
        int b   = p / (OHW * OHW);
        int rem = p - b * (OHW * OHW);
        int oh  = rem / OHW;
        int ow  = rem - oh * OHW;
        const float* base = x + (((size_t)b * INH + oh) * INW + ow) * CIN;
        #pragma unroll
        for (int kh = 0; kh < 3; ++kh)
            #pragma unroll
            for (int kw = 0; kw < 3; ++kw) {
                const float* q = base + (kh * INW + kw) * CIN;
                #pragma unroll
                for (int c = 0; c < CIN; c += 4) {
                    float4 v = *(const float4*)(q + c);
                    s += v.x * v.x + v.y * v.y + v.z * v.z + v.w * v.w;
                }
            }
        psq[p] = s;
    }
    sr[threadIdx.x] = (p < NPIX) ? s : 0.0f;
    __syncthreads();
    #pragma unroll
    for (int t = 128; t > 0; t >>= 1) {
        if (threadIdx.x < t) sr[threadIdx.x] += sr[threadIdx.x + t];
        __syncthreads();
    }
    if (threadIdx.x == 0) psqPart[blockIdx.x] = sr[0];
}

// ---------------------------------------------------------------------------
// Kernel 3 (1 block): c_sq[f] and gamma from closed-form mean:
// sum(dist) = F*sum(psq) - 2*sum(cross) + NPIX*sum(csq)
// ---------------------------------------------------------------------------
__global__ __launch_bounds__(256) void kan_gamma_kernel(
    const float* __restrict__ wcp,
    const float* __restrict__ crossPart, int nCross,
    const float* __restrict__ psqPart,   int nPsq,
    float* __restrict__ csq,
    float* __restrict__ gamma)
{
    __shared__ double sr[256];
    const int tid = threadIdx.x;

    double myCsq = 0.0;
    if (tid < NFILT) {
        float s = 0.0f;
        for (int k = 0; k < 288; ++k) {
            float v = wcp[k * NFILT + tid];
            s += v * v;
        }
        csq[tid] = s;
        myCsq = (double)s;
    }
    sr[tid] = myCsq;
    __syncthreads();
    for (int t = 128; t > 0; t >>= 1) {
        if (tid < t) sr[tid] += sr[tid + t];
        __syncthreads();
    }
    double sumCsq = sr[0];
    __syncthreads();

    double sc = 0.0;
    for (int i = tid; i < nCross; i += 256) sc += (double)crossPart[i];
    sr[tid] = sc;
    __syncthreads();
    for (int t = 128; t > 0; t >>= 1) {
        if (tid < t) sr[tid] += sr[tid + t];
        __syncthreads();
    }
    double sumCross = sr[0];
    __syncthreads();

    double sp = 0.0;
    for (int i = tid; i < nPsq; i += 256) sp += (double)psqPart[i];
    sr[tid] = sp;
    __syncthreads();
    for (int t = 128; t > 0; t >>= 1) {
        if (tid < t) sr[tid] += sr[tid + t];
        __syncthreads();
    }
    if (tid == 0) {
        double sumPsq = sr[0];
        double total  = (double)NFILT * sumPsq - 2.0 * sumCross
                      + (double)NPIX * sumCsq;
        double meanD  = total / ((double)NPIX * (double)NFILT);
        gamma[0] = (float)(1.0 / (2.0 * meanD));
    }
}

// ---------------------------------------------------------------------------
// Kernel 4: out = conv + exp(-gamma*(psq - 2*cross + csq)) + bias (float4).
// ---------------------------------------------------------------------------
__global__ __launch_bounds__(256) void kan_final_kernel(
    const float* __restrict__ cross,
    const float* __restrict__ psq,
    const float* __restrict__ csq,
    const float* __restrict__ gamma,
    const float* __restrict__ bias,
    float* out)
{
    int i = blockIdx.x * 256 + threadIdx.x;
    if (i >= TOTAL4) return;
    const float g = gamma[0];
    int p  = i >> 4;
    int fb = (i & 15) * 4;

    float4 cv = ((const float4*)out)[i];
    float4 cr = ((const float4*)cross)[i];
    float  ps = psq[p];
    float4 cs = *(const float4*)(csq + fb);
    float4 bs = *(const float4*)(bias + fb);

    float4 r;
    r.x = cv.x + __expf(-g * (ps - 2.0f * cr.x + cs.x)) + bs.x;
    r.y = cv.y + __expf(-g * (ps - 2.0f * cr.y + cs.y)) + bs.y;
    r.z = cv.z + __expf(-g * (ps - 2.0f * cr.z + cs.z)) + bs.z;
    r.w = cv.w + __expf(-g * (ps - 2.0f * cr.w + cs.w)) + bs.w;
    ((float4*)out)[i] = r;
}

// ---------------------------------------------------------------------------
extern "C" void kernel_launch(void* const* d_in, const int* in_sizes, int n_in,
                              void* d_out, int out_size, void* d_ws, size_t ws_size,
                              hipStream_t stream)
{
    const float* x     = (const float*)d_in[0];
    const float* wconv = (const float*)d_in[1];
    const float* wcp   = (const float*)d_in[2];
    const float* bias  = (const float*)d_in[3];
    float* out = (float*)d_out;
    float* ws  = (float*)d_ws;

    const size_t NCROSS = (size_t)NPIX * NFILT;
    float* cross     = ws;                       // 50,466,816 floats
    float* psq       = cross + NCROSS;           // 788,544
    float* csq       = psq + NPIX;               // 64
    float* gamma     = csq + NFILT;              // 1
    float* crossPart = gamma + 1;                // NBLK_GEMM
    float* psqPart   = crossPart + NBLK_GEMM;    // NBLK_PSQ
    size_t used = NCROSS + NPIX + NFILT + 1 + NBLK_GEMM + NBLK_PSQ;
    used = (used + 3) & ~(size_t)3;              // 16B align
    unsigned short* bws = (unsigned short*)(ws + used);   // 9*4096 bf16
    unsigned short* xb  = bws + KTAPS * BTILE_ELEMS;      // NXEL bf16 (16B-aligned:
                                                          // 36864 elems = 72KB)

    kan_prepx_kernel<<<NXUNITS / 256, 256, 0, stream>>>(x, xb);
    kan_prepb_kernel<<<(KTAPS * BTILE_ELEMS + 255) / 256, 256, 0, stream>>>(wconv, wcp, bws);
    kan_gemm_kernel<<<NBLK_GEMM, 256, 0, stream>>>(xb, bws, out, cross, crossPart);
    kan_psq_kernel<<<NBLK_PSQ, 256, 0, stream>>>(x, psq, psqPart);
    kan_gamma_kernel<<<1, 256, 0, stream>>>(wcp, crossPart, NBLK_GEMM,
                                            psqPart, NBLK_PSQ, csq, gamma);
    kan_final_kernel<<<(TOTAL4 + 255) / 256, 256, 0, stream>>>(cross, psq, csq,
                                                               gamma, bias, out);
}
